// GNNEdgeAttr_32641751449988
// MI455X (gfx1250) — matrix operand order repaired and checked by reading, not compiled, not run on hardware
//
#include <hip/hip_runtime.h>
#include <math.h>

#define NN   10000
#define EE   320000
#define FW   256      // feature width (H*C) at every layer
#define ED   64       // edge dim

typedef __attribute__((ext_vector_type(16))) __bf16 v16bf;
typedef __attribute__((ext_vector_type(8)))  float  v8f;

struct BFrag { union { uint4 u[2]; v16bf v; }; };

__device__ __forceinline__ __bf16 f2bf(float f) {
  unsigned u = __float_as_uint(f);
  u += 0x7FFFu + ((u >> 16) & 1u);          // round-to-nearest-even
  unsigned short s = (unsigned short)(u >> 16);
  __bf16 r; __builtin_memcpy(&r, &s, 2); return r;
}
__device__ __forceinline__ float bf2f(__bf16 b) {
  unsigned short s; __builtin_memcpy(&s, &b, 2);
  return __uint_as_float(((unsigned)s) << 16);
}

// ---------------------------------------------------------------- utilities
__global__ void k_f32_to_bf16(const float* __restrict__ src, __bf16* __restrict__ dst, int n) {
  int t = blockIdx.x * blockDim.x + threadIdx.x;
  if (t < n) dst[t] = f2bf(src[t]);
}
// src [K,Nc] row-major -> dst [Nc,K] row-major (bf16)
__global__ void k_transpose_bf16(const float* __restrict__ src, __bf16* __restrict__ dst,
                                 int K, int Nc) {
  int t = blockIdx.x * blockDim.x + threadIdx.x;
  if (t >= K * Nc) return;
  int n = t / K, k = t - n * K;
  dst[t] = f2bf(src[(size_t)k * Nc + n]);
}
__global__ void k_fill_u32(unsigned* __restrict__ p, unsigned v, int n) {
  int t = blockIdx.x * blockDim.x + threadIdx.x;
  if (t < n) p[t] = v;
}

// ---------------------------------------------------------------- WMMA GEMM
// C[M,Nc] = A[M,K](bf16) * Bt[Nc,K](bf16, pre-transposed B) (+bias)
// one 16x64 output tile per wave32 (4 accumulators): the A fragment is loaded
// once per 32-wide K step and reused for 4 WMMAs -> 2.5 b128 loads per wmma.
__global__ __launch_bounds__(128)
void k_gemm_bf16_wmma(const __bf16* __restrict__ A, int lda,
                      const __bf16* __restrict__ Bt, int ldb,
                      const float* __restrict__ bias,
                      float* __restrict__ Cf, int beta,
                      __bf16* __restrict__ Cb, int ldc,
                      int M, int Nc, int K) {
  const int lane = threadIdx.x & 31;
  const int tile = blockIdx.x * (blockDim.x >> 5) + (threadIdx.x >> 5);
  const int ntn  = Nc >> 6;                        // 64-wide column tiles
  if (tile >= (M >> 4) * ntn) return;
  const int m0  = (tile / ntn) << 4;
  const int n0  = (tile % ntn) << 6;
  const int l15 = lane & 15;
  // A frag: lane(0-15)->row, K elems {kb..kb+7, kb+16..kb+23}, kb = 8*(lane>>4)
  const __bf16* ap = A  + (size_t)(m0 + l15) * lda + ((lane >> 4) << 3);
  // B frag: lane(0-15)->col (row of Bt), K elems {kb..kb+7, kb+16..kb+23}, kb = 8*(lane>>4)
  const __bf16* bp = Bt + (size_t)(n0 + l15) * ldb + ((lane >> 4) << 3);
  const size_t bstep = (size_t)16 * ldb;           // next 16-column sub-tile
  v8f acc[4];
#pragma unroll
  for (int s = 0; s < 4; ++s) acc[s] = (v8f){0.f, 0.f, 0.f, 0.f, 0.f, 0.f, 0.f, 0.f};
  for (int k0 = 0; k0 < K; k0 += 32) {
    BFrag a;
    a.u[0] = *(const uint4*)(ap + k0);
    a.u[1] = *(const uint4*)(ap + k0 + 16);
    BFrag b[4];
#pragma unroll
    for (int s = 0; s < 4; ++s) {
      b[s].u[0] = *(const uint4*)(bp + s * bstep + k0);
      b[s].u[1] = *(const uint4*)(bp + s * bstep + k0 + 16);
    }
#pragma unroll
    for (int s = 0; s < 4; ++s)
      acc[s] = __builtin_amdgcn_wmma_f32_16x16x32_bf16(false, a.v, false, b[s].v,
                                                       (short)0, acc[s], false, false);
  }
  const int rb = m0 + ((lane >> 4) << 3);
#pragma unroll
  for (int s = 0; s < 4; ++s) {
    const int col = n0 + s * 16 + l15;
    const float bv = bias ? bias[col] : 0.f;
    for (int r = 0; r < 8; ++r) {
      float x = acc[s][r] + bv;
      size_t idx = (size_t)(rb + r) * ldc + col;
      if (Cf) Cf[idx] = beta ? (Cf[idx] + x) : x;
      if (Cb) Cb[idx] = f2bf(x);
    }
  }
}

// ---------------------------------------------------------------- attention
// pass 1: alpha[e,h] = scale * ( q[dst]·k[src] + edge_attr[e]·G[dst,h] ), segment max
// one wave per edge; lane handles 8 contiguous channels; head = contiguous lane groups
__global__ __launch_bounds__(128)
void k_attn_alpha(const int* __restrict__ src, const int* __restrict__ dst,
                  const __bf16* __restrict__ Q, const __bf16* __restrict__ Kb,
                  const __bf16* __restrict__ G, const __bf16* __restrict__ EA,
                  float* __restrict__ alpha, unsigned* __restrict__ amaxKey,
                  int E, int H, float scale) {
  int e = blockIdx.x * (blockDim.x >> 5) + (threadIdx.x >> 5);
  if (e >= E) return;
  const int lane = threadIdx.x & 31;
  const int s = src[e], d = dst[e];
  union H8 { uint4 u; __bf16 h[8]; } q8, k8, g8, a8;
  q8.u = *(const uint4*)(Q  + (size_t)d * FW + 8 * lane);
  k8.u = *(const uint4*)(Kb + (size_t)s * FW + 8 * lane);
  float part = 0.f;
#pragma unroll
  for (int j = 0; j < 8; ++j) part += bf2f(q8.h[j]) * bf2f(k8.h[j]);
  if (lane < 8 * H) {                     // edge term: width H*64
    g8.u = *(const uint4*)(G  + (size_t)d * (H * 64) + 8 * lane);
    a8.u = *(const uint4*)(EA + (size_t)e * ED + ((8 * lane) & 63));
#pragma unroll
    for (int j = 0; j < 8; ++j) part += bf2f(g8.h[j]) * bf2f(a8.h[j]);
  }
  const int g = 32 / H;                   // lanes per head
  for (int off = g >> 1; off; off >>= 1) part += __shfl_xor(part, off, 32);
  if ((lane & (g - 1)) == 0) {
    int h = lane / g;
    float a = part * scale;
    alpha[(size_t)e * H + h] = a;
    unsigned ub  = __float_as_uint(a);
    unsigned key = (ub & 0x80000000u) ? ~ub : (ub | 0x80000000u);  // order-preserving
    atomicMax(&amaxKey[d * H + h], key);
  }
}

// pass 2: ea = exp(alpha - amax[dst]); denom[dst] += ea   (ea stored over alpha)
__global__ void k_attn_expsum(const int* __restrict__ dst, float* __restrict__ alpha,
                              const unsigned* __restrict__ amaxKey,
                              float* __restrict__ denom, int E, int H) {
  int t = blockIdx.x * blockDim.x + threadIdx.x;
  if (t >= E * H) return;
  int e = t / H, h = t - e * H;
  int d = dst[e];
  unsigned key  = amaxKey[d * H + h];
  unsigned bits = (key & 0x80000000u) ? (key ^ 0x80000000u) : ~key;
  float mx = __uint_as_float(bits);
  float ea = __expf(alpha[t] - mx);
  alpha[t] = ea;
  atomicAdd(&denom[d * H + h], ea);
}

// pass 3: outv[dst] += w * v[src];  z[dst,h,0:64] += w * edge_attr[e]
__global__ __launch_bounds__(128)
void k_attn_aggregate(const int* __restrict__ src, const int* __restrict__ dst,
                      const __bf16* __restrict__ V, const __bf16* __restrict__ EA,
                      const float* __restrict__ ea, const float* __restrict__ denom,
                      float* __restrict__ outv, float* __restrict__ z,
                      int E, int H, int C) {
  int e = blockIdx.x * (blockDim.x >> 5) + (threadIdx.x >> 5);
  if (e >= E) return;
  const int lane = threadIdx.x & 31;
  const int s = src[e], d = dst[e];
  if (e + 256 < E)                         // stream-ahead hint for edge-attr rows
    __builtin_prefetch(EA + (size_t)(e + 256) * ED, 0, 1);
  union H8 { uint4 u; __bf16 h[8]; } v8, a8;
  const int hv = (lane * 8) / C;          // head owning these v-channels
  float wv = ea[(size_t)e * H + hv] / (denom[d * H + hv] + 1e-16f);
  v8.u = *(const uint4*)(V + (size_t)s * FW + 8 * lane);
#pragma unroll
  for (int j = 0; j < 8; ++j)
    atomicAdd(&outv[(size_t)d * FW + 8 * lane + j], wv * bf2f(v8.h[j]));
  if (lane < 8 * H) {
    const int hz = lane >> 3;
    float wz = ea[(size_t)e * H + hz] / (denom[d * H + hz] + 1e-16f);
    a8.u = *(const uint4*)(EA + (size_t)e * ED + ((8 * lane) & 63));
#pragma unroll
    for (int j = 0; j < 8; ++j)
      atomicAdd(&z[(size_t)d * (H * 64) + 8 * lane + j], wz * bf2f(a8.h[j]));
  }
}

// epilogue: res = outv + skip; optional relu; -> bf16 (next layer) or f32 (final out)
__global__ void k_epilogue(const float* __restrict__ outv, const float* __restrict__ skip,
                           __bf16* __restrict__ hb, float* __restrict__ fo,
                           int n, int relu) {
  int t = blockIdx.x * blockDim.x + threadIdx.x;
  if (t >= n) return;
  float x = outv[t] + skip[t];
  if (relu) x = fmaxf(x, 0.f);
  if (hb) hb[t] = f2bf(x);
  if (fo) fo[t] = x;
}

// ---------------------------------------------------------------- host
extern "C" void kernel_launch(void* const* d_in, const int* in_sizes, int n_in,
                              void* d_out, int out_size, void* d_ws, size_t ws_size,
                              hipStream_t stream) {
  const float* x   = (const float*)d_in[0];
  const int*   ei  = (const int*)d_in[1];        // [2,E] int32
  const float* eat = (const float*)d_in[2];
  const float* P[3][9];                           // Wq,bq,Wk,bk,Wv,bv,We,Ws,bs
  for (int l = 0; l < 3; ++l)
    for (int j = 0; j < 9; ++j) P[l][j] = (const float*)d_in[3 + l * 9 + j];
  const int* srcIdx = ei;
  const int* dstIdx = ei + EE;

  // ---- workspace carve-up (256B aligned)
  char* base = (char*)d_ws; size_t off = 0;
  auto alloc = [&](size_t bytes) -> char* {
    char* p = base + off; off += (bytes + 255) & ~(size_t)255; return p;
  };
  __bf16 *wqt[3], *wkt[3], *wvt[3], *wst[3], *weo[3], *wet[3];
  for (int l = 0; l < 3; ++l) {
    wqt[l] = (__bf16*)alloc(65536 * 2); wkt[l] = (__bf16*)alloc(65536 * 2);
    wvt[l] = (__bf16*)alloc(65536 * 2); wst[l] = (__bf16*)alloc(65536 * 2);
    weo[l] = (__bf16*)alloc(16384 * 2); wet[l] = (__bf16*)alloc(16384 * 2);
  }
  __bf16* hA  = (__bf16*)alloc((size_t)NN * FW * 2);
  __bf16* hB  = (__bf16*)alloc((size_t)NN * FW * 2);
  __bf16* Qb  = (__bf16*)alloc((size_t)NN * FW * 2);
  __bf16* Kb  = (__bf16*)alloc((size_t)NN * FW * 2);
  __bf16* Vb  = (__bf16*)alloc((size_t)NN * FW * 2);
  __bf16* Gb  = (__bf16*)alloc((size_t)NN * FW * 2);
  __bf16* eab = (__bf16*)alloc((size_t)EE * ED * 2);
  __bf16* zb  = (__bf16*)alloc((size_t)NN * FW * 2);
  float*  skp = (float*)alloc((size_t)NN * FW * 4);
  float*  alp = (float*)alloc((size_t)EE * 4 * 4);
  unsigned* amx = (unsigned*)alloc((size_t)NN * 4 * 4);
  float*  den = (float*)alloc((size_t)NN * 4 * 4);
  float*  ov  = (float*)alloc((size_t)NN * FW * 4);
  float*  zz  = (float*)alloc((size_t)NN * FW * 4);

  auto gemm = [&](const __bf16* A, int lda, const __bf16* Bt, int ldb,
                  const float* bias, float* Cf, int beta, __bf16* Cb, int ldc,
                  int M, int Nc, int K) {
    int tiles = (M >> 4) * (Nc >> 6);             // 16x64 tiles
    k_gemm_bf16_wmma<<<(tiles + 3) / 4, 128, 0, stream>>>(A, lda, Bt, ldb, bias,
                                                          Cf, beta, Cb, ldc, M, Nc, K);
  };
  auto fill0 = [&](void* p, int n) {
    k_fill_u32<<<(n + 255) / 256, 256, 0, stream>>>((unsigned*)p, 0u, n);
  };

  // ---- weight prep: bf16 + pre-transposed B for the WMMA GEMM
  for (int l = 0; l < 3; ++l) {
    k_transpose_bf16<<<(65536 + 255) / 256, 256, 0, stream>>>(P[l][0], wqt[l], 256, 256);
    k_transpose_bf16<<<(65536 + 255) / 256, 256, 0, stream>>>(P[l][2], wkt[l], 256, 256);
    k_transpose_bf16<<<(65536 + 255) / 256, 256, 0, stream>>>(P[l][4], wvt[l], 256, 256);
    k_transpose_bf16<<<(65536 + 255) / 256, 256, 0, stream>>>(P[l][7], wst[l], 256, 256);
    k_f32_to_bf16   <<<(16384 + 255) / 256, 256, 0, stream>>>(P[l][6], weo[l], 16384);
    k_transpose_bf16<<<(16384 + 255) / 256, 256, 0, stream>>>(P[l][6], wet[l], 64, 256);
  }
  k_f32_to_bf16<<<(NN * FW + 255) / 256, 256, 0, stream>>>(x,   hA,  NN * FW);
  k_f32_to_bf16<<<(EE * ED + 255) / 256, 256, 0, stream>>>(eat, eab, EE * ED);

  const __bf16* hcur = hA; __bf16* hnext = hB;
  const int eblocks = (EE + 3) / 4;

  for (int L = 0; L < 3; ++L) {
    const int H = (L < 2) ? 4 : 1;
    const int C = (L < 2) ? 64 : 256;
    const float scale = 1.f / sqrtf((float)C);

    // node projections (WMMA): Q/K/V (bf16 out, bias) + skip (f32 out, bias)
    gemm(hcur, FW, wqt[L], 256, P[L][1], nullptr, 0, Qb, FW, NN, FW, 256);
    gemm(hcur, FW, wkt[L], 256, P[L][3], nullptr, 0, Kb, FW, NN, FW, 256);
    gemm(hcur, FW, wvt[L], 256, P[L][5], nullptr, 0, Vb, FW, NN, FW, 256);
    gemm(hcur, FW, wst[L], 256, P[L][8], skp, 0, nullptr, FW, NN, FW, 256);
    // G[n,h,k] = sum_c We[k,h*C+c] * Q[n,h*C+c]  (per-head WMMA GEMM)
    for (int h = 0; h < H; ++h)
      gemm(Qb + h * C, FW, weo[L] + h * C, 256, nullptr, nullptr, 0,
           Gb + h * 64, H * 64, NN, 64, C);

    fill0(amx, NN * H); fill0(den, NN * H);
    fill0(ov, NN * FW); fill0(zz, NN * H * 64);

    k_attn_alpha<<<eblocks, 128, 0, stream>>>(srcIdx, dstIdx, Qb, Kb, Gb, eab,
                                              alp, amx, EE, H, scale);
    k_attn_expsum<<<(EE * H + 255) / 256, 256, 0, stream>>>(dstIdx, alp, amx, den, EE, H);
    k_attn_aggregate<<<eblocks, 128, 0, stream>>>(srcIdx, dstIdx, Vb, eab, alp, den,
                                                  ov, zz, EE, H, C);
    // fold edge-attr contribution: outv[:,h*64:...] += z_h @ We_h   (WMMA, beta=1)
    k_f32_to_bf16<<<(NN * H * 64 + 255) / 256, 256, 0, stream>>>(zz, zb, NN * H * 64);
    for (int h = 0; h < H; ++h)
      gemm(zb + h * 64, H * 64, wet[L] + h * 64 * 64, 64, nullptr,
           ov + h * 64, 1, nullptr, FW, NN, C, 64);

    k_epilogue<<<(NN * FW + 255) / 256, 256, 0, stream>>>(
        ov, skp, (L < 2) ? hnext : nullptr, (L < 2) ? nullptr : (float*)d_out,
        NN * FW, (L < 2) ? 1 : 0);
    const __bf16* t = hcur; hcur = hnext; hnext = (__bf16*)t;
  }
  (void)in_sizes; (void)n_in; (void)out_size; (void)ws_size;
}
